// SCANN_5446018531524
// MI455X (gfx1250) — matrix operand order repaired and checked by reading, not compiled, not run on hardware
//
#include <hip/hip_runtime.h>
#include <hip/hip_bf16.h>

typedef _Float16     v16h  __attribute__((ext_vector_type(16)));
typedef float        v8f   __attribute__((ext_vector_type(8)));
typedef unsigned int u32x4 __attribute__((ext_vector_type(4)));
typedef int          i32x4 __attribute__((ext_vector_type(4)));
typedef int          i32x8 __attribute__((ext_vector_type(8)));

#define IMGD 127
#define SPND 64
#define PND  4096
#define GN   64
#define BN   16

// ---------------------------------------------------------------------------
// Pass 1: per-(b,g) augmented Gram matrix  M_aug = t_aug^T * t_aug  (10x10,
// padded to 16x16) via V_WMMA_F32_16X16X32_F16.  One block per (b,g).
// The block's whole 127x127 fp32 plane (64.5 KB) is DMA'd into LDS once by
// the Tensor Data Mover (TENSOR_LOAD_TO_LDS, waited with s_wait_tensorcnt),
// then 4 waves each reduce 1024 patches in K=32 chunks from LDS; LDS float
// atomics combine the per-wave accumulators.
// ---------------------------------------------------------------------------
__global__ __launch_bounds__(128) void scann_gram_kernel(
    const float* __restrict__ x, float* __restrict__ gram) {
  __shared__ float    planeLds[IMGD * IMGD];   // 64516 B, staged by TDM
  __shared__ _Float16 ldsT[4][32][16];         // [wave][patch-in-chunk][comp]
  __shared__ float    ldsAcc[256];

  const int bg = blockIdx.x;                   // b*64+g == channel plane index
  const float* plane = x + bg * (IMGD * IMGD);

  // --- TDM: DMA the whole plane into LDS (wave 0 issues one descriptor) ----
  if (threadIdx.x < 32) {
    const unsigned long long ga = (unsigned long long)(size_t)plane;
    const unsigned int ldsa = (unsigned int)(size_t)planeLds; // addr[31:0] = LDS offset
    const unsigned int N = IMGD * IMGD;        // 16129 elements, 1D tensor

    u32x4 g0 = {};
    g0[0] = 1u;                                         // count=1 (valid D#)
    g0[1] = ldsa;                                       // lds_addr
    g0[2] = (unsigned int)(ga & 0xFFFFFFFFu);           // global_addr[31:0]
    g0[3] = (unsigned int)((ga >> 32) & 0x01FFFFFFu)    // global_addr[56:32]
          | (2u << 30);                                 // type = 2 ("image")

    i32x8 g1 = {};
    g1[0] = (int)(2u << 16);                            // data_size = 4 bytes
    g1[1] = (int)((N & 0xFFFFu) << 16);                 // tensor_dim0[15:0]
    g1[2] = (int)((N >> 16) | (1u << 16));              // tensor_dim0[31:16], tensor_dim1=1
    g1[3] = (int)((N & 0xFFFFu) << 16);                 // tile_dim0 = N
    g1[4] = 0;                                          // tile_dim1=0 -> 1D
    g1[5] = (int)N;                                     // tensor_dim0_stride
    g1[6] = 0;
    g1[7] = 0;

    i32x4 g2 = {};                                      // <=2D: groups 2/3 unused
    i32x4 g3 = {};
#if defined(__clang_major__) && (__clang_major__ >= 23)
    i32x8 gz = {};
    __builtin_amdgcn_tensor_load_to_lds(g0, g1, g2, g3, gz, 0);
#else
    __builtin_amdgcn_tensor_load_to_lds(g0, g1, g2, g3, 0);
#endif
    __builtin_amdgcn_s_wait_tensorcnt(0);
  }
  __syncthreads();                             // publish plane to all waves

  const int lane = threadIdx.x & 31;
  const int wave = threadIdx.x >> 5;
  const int lrow = lane & 15;                  // matrix row (A) / column (B)
  const int hsel = lane >> 4;                  // lane-half selector

  v8f c = {};                                  // f32 accumulator (16x16)
  const int waveBase = wave * 1024;

  for (int it = 0; it < 32; ++it) {
    // --- each lane builds one patch (9 taps, zero-padded borders) from LDS --
    const int p   = waveBase + it * 32 + lane;
    const int py  = p >> 6, px = p & 63;
    const int iy0 = py * 2 - 1, ix0 = px * 2 - 1;
    _Float16 vals[16];
#pragma unroll
    for (int kh = 0; kh < 3; ++kh) {
#pragma unroll
      for (int kw = 0; kw < 3; ++kw) {
        const int iy = iy0 + kh, ix = ix0 + kw;
        const bool ok = (iy >= 0) & (iy < IMGD) & (ix >= 0) & (ix < IMGD);
        const int cy = iy < 0 ? 0 : (iy > IMGD - 1 ? IMGD - 1 : iy);
        const int cx = ix < 0 ? 0 : (ix > IMGD - 1 ? IMGD - 1 : ix);
        const float v = planeLds[cy * IMGD + cx];
        vals[kh * 3 + kw] = (_Float16)(ok ? v : 0.0f);
      }
    }
    vals[9] = (_Float16)1.0f;                  // augmentation -> col sums & L
#pragma unroll
    for (int u = 10; u < 16; ++u) vals[u] = (_Float16)0.0f;
#pragma unroll
    for (int u = 0; u < 16; ++u) ldsT[wave][lane][u] = vals[u];
    __syncthreads();

    // --- build WMMA fragments (ISA 7.12.2 16-bit layouts) -------------------
    v16h a, b;
#pragma unroll
    for (int e = 0; e < 8; ++e) {              // A[m=lrow, k]
      a[e]     = ldsT[wave][hsel * 8 + e][lrow];
      a[8 + e] = ldsT[wave][16 + hsel * 8 + e][lrow];
    }
#pragma unroll
    for (int e = 0; e < 16; ++e) {             // B[k, n=lrow]
      b[e] = ldsT[wave][e < 8 ? hsel * 8 + e : 16 + hsel * 8 + (e - 8)][lrow];
    }
    c = __builtin_amdgcn_wmma_f32_16x16x32_f16(false, a, false, b,
                                               (short)0, c, false, false);
    __syncthreads();
  }

  // --- cross-wave reduction in LDS -----------------------------------------
  for (int i = threadIdx.x; i < 256; i += 128) ldsAcc[i] = 0.0f;
  __syncthreads();
#pragma unroll
  for (int j = 0; j < 8; ++j) {                // C: vgpr j -> M = j + 8*hsel
    atomicAdd(&ldsAcc[(j + 8 * hsel) * 16 + lrow], c[j]);
  }
  __syncthreads();
  float* o = gram + bg * 256;
  for (int i = threadIdx.x; i < 256; i += 128) o[i] = ldsAcc[i];
}

// ---------------------------------------------------------------------------
// Pass 2: per-(b,g) thread: E = Wk_aug^T * M_aug * Wq_aug, row softmax,
// collapse to 9-tap effective filter weff + bias beff.
// ---------------------------------------------------------------------------
__global__ __launch_bounds__(256) void scann_attn_kernel(
    const float* __restrict__ gram,
    const float* __restrict__ Wq, const float* __restrict__ bq,
    const float* __restrict__ Wk, const float* __restrict__ bk,
    const float* __restrict__ Wv, const float* __restrict__ bv,
    const float* __restrict__ Wo, const float* __restrict__ bo,
    float* __restrict__ eff) {
  const int idx = blockIdx.x * 256 + threadIdx.x;
  if (idx >= BN * GN) return;
  const int g = idx & (GN - 1);
  const float* M = gram + idx * 256;

  float Ma[10][10];
#pragma unroll
  for (int u = 0; u < 10; ++u)
#pragma unroll
    for (int v = 0; v < 10; ++v) Ma[u][v] = M[u * 16 + v];

  const float* wq = Wq + g * 81; const float* bqv = bq + g * 9;
  const float* wk = Wk + g * 81; const float* bkv = bk + g * 9;

  float E[9][9];
#pragma unroll
  for (int t = 0; t < 9; ++t) {
    float ct[10];
#pragma unroll
    for (int u = 0; u < 10; ++u) {
      float acc = 0.0f;
#pragma unroll
      for (int v = 0; v < 10; ++v) {
        const float w = (v < 9) ? wq[v * 9 + t] : bqv[t];  // Wq_aug[v][t]
        acc += Ma[u][v] * w;
      }
      ct[u] = acc;
    }
#pragma unroll
    for (int s = 0; s < 9; ++s) {
      float e = 0.0f;
#pragma unroll
      for (int u = 0; u < 10; ++u) {
        const float w = (u < 9) ? wk[u * 9 + s] : bkv[s];  // Wk_aug[u][s]
        e += w * ct[u];
      }
      E[s][t] = e;
    }
  }

  float w2[9];
#pragma unroll
  for (int t = 0; t < 9; ++t) w2[t] = 0.0f;
#pragma unroll
  for (int s = 0; s < 9; ++s) {
    float mx = E[s][0];
#pragma unroll
    for (int t = 1; t < 9; ++t) mx = fmaxf(mx, E[s][t]);
    float ex[9], sum = 0.0f;
#pragma unroll
    for (int t = 0; t < 9; ++t) { ex[t] = __expf(E[s][t] - mx); sum += ex[t]; }
    const float scale = Wo[g * 9 + s] / sum;
#pragma unroll
    for (int t = 0; t < 9; ++t) w2[t] += ex[t] * scale;  // w2 = attn^T * Wo
  }

  const float* wv = Wv + g * 81; const float* bvv = bv + g * 9;
  float* o = eff + idx * 16;
  float beff = bo[g];
#pragma unroll
  for (int u = 0; u < 9; ++u) {
    float acc = 0.0f;
#pragma unroll
    for (int t = 0; t < 9; ++t) acc += wv[u * 9 + t] * w2[t];
    o[u] = acc;                                // weff[u]
  }
#pragma unroll
  for (int t = 0; t < 9; ++t) beff += bvv[t] * w2[t];
  o[9] = beff;
}

// ---------------------------------------------------------------------------
// Pass 3: apply the per-(b,g) effective 3x3 stride-2 filter.
// ---------------------------------------------------------------------------
__global__ __launch_bounds__(256) void scann_conv_kernel(
    const float* __restrict__ x, const float* __restrict__ eff,
    float* __restrict__ out) {
  const int tid = blockIdx.x * 256 + threadIdx.x;
  if (tid >= BN * GN * PND) return;
  const int bg = tid >> 12;
  const int l  = tid & (PND - 1);
  const int py = l >> 6, px = l & 63;
  const float* plane = x + bg * (IMGD * IMGD);
  const float* e = eff + bg * 16;
  const int iy0 = py * 2 - 1, ix0 = px * 2 - 1;
  float acc = e[9];
#pragma unroll
  for (int kh = 0; kh < 3; ++kh) {
#pragma unroll
    for (int kw = 0; kw < 3; ++kw) {
      const int iy = iy0 + kh, ix = ix0 + kw;
      const bool ok = (iy >= 0) & (iy < IMGD) & (ix >= 0) & (ix < IMGD);
      const int cy = iy < 0 ? 0 : (iy > IMGD - 1 ? IMGD - 1 : iy);
      const int cx = ix < 0 ? 0 : (ix > IMGD - 1 ? IMGD - 1 : ix);
      const float v = plane[cy * IMGD + cx];
      acc += (ok ? v : 0.0f) * e[kh * 3 + kw];
    }
  }
  out[tid] = acc;
}

extern "C" void kernel_launch(void* const* d_in, const int* in_sizes, int n_in,
                              void* d_out, int out_size, void* d_ws, size_t ws_size,
                              hipStream_t stream) {
  const float* x  = (const float*)d_in[0];
  const float* Wq = (const float*)d_in[1];
  const float* bq = (const float*)d_in[2];
  const float* Wk = (const float*)d_in[3];
  const float* bk = (const float*)d_in[4];
  const float* Wv = (const float*)d_in[5];
  const float* bv = (const float*)d_in[6];
  const float* Wo = (const float*)d_in[7];
  const float* bo = (const float*)d_in[8];

  float* gram = (float*)d_ws;                  // 1024 * 256 floats = 1 MB
  float* eff  = gram + BN * GN * 256;          // 1024 * 16 floats

  scann_gram_kernel<<<BN * GN, 128, 0, stream>>>(x, gram);
  scann_attn_kernel<<<(BN * GN + 255) / 256, 256, 0, stream>>>(
      gram, Wq, bq, Wk, bk, Wv, bv, Wo, bo, eff);
  scann_conv_kernel<<<(BN * GN * PND + 255) / 256, 256, 0, stream>>>(
      x, eff, (float*)d_out);
}
